// BasicTransformerBlock_23656679867598
// MI455X (gfx1250) — matrix operand order repaired and checked by reading, not compiled, not run on hardware
//
#include <hip/hip_runtime.h>
#include <hip/hip_bf16.h>

typedef __bf16 bf16;
typedef __bf16 v16bf __attribute__((ext_vector_type(16)));
typedef float  v8f   __attribute__((ext_vector_type(8)));

#define WMMA_BF16(A, B, C) \
  __builtin_amdgcn_wmma_f32_16x16x32_bf16(false, (A), false, (B), (short)0, (C), false, false)

#if __has_builtin(__builtin_amdgcn_tensor_load_to_lds) && \
    __has_builtin(__builtin_amdgcn_s_wait_tensorcnt)
#define HAVE_TDM 1
#else
#define HAVE_TDM 0
#endif

#if HAVE_TDM
typedef unsigned int tdm_u32x4 __attribute__((ext_vector_type(4)));
typedef int          tdm_i32x8 __attribute__((ext_vector_type(8)));
typedef int          tdm_i32x4 __attribute__((ext_vector_type(4)));

// ---------------------------------------------------------------------------
// TDM 2D tile load: global (row-major, row_stride elems of bf16) -> LDS
// (contiguous tile rows). Rows beyond valid_rows are hardware zero-filled
// (OOB reads return 0 per CDNA5 ISA). D# packing per ISA p.152-156.
// Issued per-wave (EXEC ignored): call from ONE wave only. Tracked: TENSORcnt.
// This toolchain uses the 6-arg builtin (extra i32x8 group before cpol).
// ---------------------------------------------------------------------------
__device__ __forceinline__ void tdm_load_tile(unsigned lds_byte_off, const bf16* gtile,
                                              unsigned tile_cols, unsigned tile_rows,
                                              int valid_rows, unsigned row_stride) {
  unsigned long long ga = (unsigned long long)(uintptr_t)gtile;
  unsigned td0 = tile_cols;                                   // tensor_dim0 (elems)
  unsigned td1 = (unsigned)(valid_rows > 0 ? valid_rows : 0); // tensor_dim1 (rows)
  unsigned long long st0 = (unsigned long long)row_stride;    // tensor_dim0_stride
  tdm_u32x4 g0;
  g0[0] = 1u;                                        // count=1 (user descriptor)
  g0[1] = lds_byte_off;                              // lds_addr
  g0[2] = (unsigned)(ga & 0xffffffffu);              // global_addr[31:0]
  g0[3] = (unsigned)((ga >> 32) & 0x01ffffffu)       // global_addr[56:32]
        | (2u << 30);                                // type=2 ("image")
  tdm_i32x8 g1;
  g1[0] = (int)(1u << 16);                                           // data_size=1 (2B)
  g1[1] = (int)((td0 & 0xffffu) << 16);                              // tensor_dim0 lo
  g1[2] = (int)(((td0 >> 16) & 0xffffu) | ((td1 & 0xffffu) << 16));  // td0 hi | td1 lo
  g1[3] = (int)(((td1 >> 16) & 0xffffu) | ((tile_cols & 0xffffu) << 16)); // td1 hi | tile_dim0
  g1[4] = (int)(tile_rows & 0xffffu);                                // tile_dim1 (tile_dim2=0)
  g1[5] = (int)(st0 & 0xffffffffu);                                  // dim0_stride lo
  g1[6] = (int)((st0 >> 32) & 0xffffu);                              // dim0_stride hi
  g1[7] = 0;                                                         // dim1_stride=0 (2D)
  tdm_i32x4 z4 = {0, 0, 0, 0};
  tdm_i32x8 z8 = {0, 0, 0, 0, 0, 0, 0, 0};
  __builtin_amdgcn_tensor_load_to_lds(g0, g1, z4, z4, z8, 0);
}
#define LDS_OFF(p) ((unsigned)(uintptr_t)(p))
#endif

// ---------------------------------------------------------------------------
// fp32 -> bf16 streaming convert
// ---------------------------------------------------------------------------
__global__ void cvt_f32_bf16(const float* __restrict__ in, bf16* __restrict__ out, int n) {
  int i = blockIdx.x * blockDim.x + threadIdx.x;
  if (i < n) out[i] = (bf16)in[i];
}

// ---------------------------------------------------------------------------
// LayerNorm: one row per block (D=1024), fp32 in -> bf16 out
// ---------------------------------------------------------------------------
__global__ void ln_bf16(const float* __restrict__ x, const float* __restrict__ g,
                        const float* __restrict__ b, bf16* __restrict__ out, int D) {
  __shared__ float s1[256];
  __shared__ float s2[256];
  int row = blockIdx.x, t = threadIdx.x;
  const float* xr = x + (size_t)row * D;
  float sum = 0.f, sq = 0.f;
  for (int i = t; i < D; i += 256) { float v = xr[i]; sum += v; sq += v * v; }
  s1[t] = sum; s2[t] = sq; __syncthreads();
  for (int o = 128; o > 0; o >>= 1) {
    if (t < o) { s1[t] += s1[t + o]; s2[t] += s2[t + o]; }
    __syncthreads();
  }
  float mu  = s1[0] / (float)D;
  float var = s2[0] / (float)D - mu * mu;
  float rs  = rsqrtf(var + 1e-5f);
  bf16* orow = out + (size_t)row * D;
  for (int i = t; i < D; i += 256) orow[i] = (bf16)((xr[i] - mu) * rs * g[i] + b[i]);
}

// ---------------------------------------------------------------------------
// Fragment builders (LDS -> VGPR), contiguous k order used by this code for both A and B:
//   A 16x32: lane(row=q, half=h): a[e] = A[row][16h+e]
//   B 32x16: lane(col=q, half=h): b[e] = B[16h+e][col]
// ---------------------------------------------------------------------------
__device__ __forceinline__ v16bf frag_a(const bf16* rowp, int h) {
  v16bf a;
  #pragma unroll
  for (int e = 0; e < 16; ++e) a[e] = rowp[16 * h + e];
  return a;
}
__device__ __forceinline__ v16bf frag_b(const bf16* rowp, int h) {
  v16bf b;
  #pragma unroll
  for (int e = 0; e < 16; ++e) b[e] = rowp[16 * h + e];
  return b;
}

// ---------------------------------------------------------------------------
// Tiled WMMA GEMM: C[M,N] = A[M,K](bf16,rm) x W[N,K](bf16,rm)^T
// Block 256 thr (8 waves). Block tile 64(M) x 128(N); wave tile 32x32.
// TDM path: double-buffered tensor_load_to_lds (wave 0 issues; DMA of tile
// i+1 overlaps WMMA on tile i; s_wait_tensorcnt(2) = older pair complete).
// ---------------------------------------------------------------------------
__global__ void gemm_bf16(const bf16* __restrict__ A, const bf16* __restrict__ W,
                          const float* __restrict__ bias, const float* __restrict__ resid,
                          float* __restrict__ outF, bf16* __restrict__ outH,
                          int M, int N, int K) {
#if HAVE_TDM
  __shared__ bf16 sA[2][64][32];
  __shared__ bf16 sW[2][128][32];
#else
  __shared__ bf16 sA1[64][32];
  __shared__ bf16 sW1[128][32];
#endif
  int t = threadIdx.x;
  int wave = t >> 5, lane = t & 31;
  int h = lane >> 4, q = lane & 15;
  int wm = wave >> 2, wn = wave & 3;
  int bm = blockIdx.y * 64, bn = blockIdx.x * 128;
  int NT = K >> 5;
  v8f acc[2][2] = {};
#if HAVE_TDM
  if (wave == 0) {
    tdm_load_tile(LDS_OFF(&sA[0][0][0]), A + (size_t)bm * K, 32, 64, M - bm, K);
    tdm_load_tile(LDS_OFF(&sW[0][0][0]), W + (size_t)bn * K, 32, 128, N - bn, K);
  }
#endif
  for (int it = 0; it < NT; ++it) {
    int k0 = it << 5;
    __syncthreads();                      // everyone done reading the other buffer
#if HAVE_TDM
    int buf = it & 1;
    if (wave == 0) {
      if (it + 1 < NT) {
        tdm_load_tile(LDS_OFF(&sA[buf ^ 1][0][0]), A + (size_t)bm * K + k0 + 32, 32, 64,  M - bm, K);
        tdm_load_tile(LDS_OFF(&sW[buf ^ 1][0][0]), W + (size_t)bn * K + k0 + 32, 32, 128, N - bn, K);
        __builtin_amdgcn_s_wait_tensorcnt(2);   // current tile's pair is done
      } else {
        __builtin_amdgcn_s_wait_tensorcnt(0);
      }
    }
    __syncthreads();                      // publish current tile to all waves
    const bf16 (*cA)[32] = sA[buf];
    const bf16 (*cW)[32] = sW[buf];
#else
    { // stage A: one uint4 (8 halves) per thread, row-uniform guard
      int idx = t * 8; int r = idx >> 5, c = idx & 31;
      int gr = bm + r;
      uint4 v = {0u, 0u, 0u, 0u};
      if (gr < M) v = *(const uint4*)(A + (size_t)gr * K + k0 + c);
      *(uint4*)(&sA1[r][c]) = v;
    }
    { // stage W: two uint4 per thread
      #pragma unroll
      for (int p = 0; p < 2; ++p) {
        int idx = t * 16 + p * 8; int r = idx >> 5, c = idx & 31;
        *(uint4*)(&sW1[r][c]) = *(const uint4*)(W + (size_t)(bn + r) * K + k0 + c);
        if (k0 + 32 < K)
          __builtin_prefetch(W + (size_t)(bn + r) * K + k0 + 32 + c, 0, 1);
      }
    }
    __syncthreads();
    const bf16 (*cA)[32] = sA1;
    const bf16 (*cW)[32] = sW1;
#endif
    v16bf a0 = frag_a(&cA[wm * 32 + q][0], h);
    v16bf a1 = frag_a(&cA[wm * 32 + 16 + q][0], h);
    v16bf b0 = frag_b(&cW[wn * 32 + q][0], h);
    v16bf b1 = frag_b(&cW[wn * 32 + 16 + q][0], h);
    acc[0][0] = WMMA_BF16(a0, b0, acc[0][0]);
    acc[0][1] = WMMA_BF16(a0, b1, acc[0][1]);
    acc[1][0] = WMMA_BF16(a1, b0, acc[1][0]);
    acc[1][1] = WMMA_BF16(a1, b1, acc[1][1]);
  }
  // C/D layout: lane col = q, row = r + 8h
  #pragma unroll
  for (int tm = 0; tm < 2; ++tm) {
    #pragma unroll
    for (int tn = 0; tn < 2; ++tn) {
      #pragma unroll
      for (int r = 0; r < 8; ++r) {
        int m = bm + wm * 32 + tm * 16 + r + 8 * h;
        int n = bn + wn * 32 + tn * 16 + q;
        if (m < M) {
          float v = acc[tm][tn][r];
          if (bias)  v += bias[n];
          if (resid) v += resid[(size_t)m * N + n];
          if (outF) outF[(size_t)m * N + n] = v;
          else      outH[(size_t)m * N + n] = (bf16)v;
        }
      }
    }
  }
}

// ---------------------------------------------------------------------------
// GEGLU: out[m,n] = (A·Wh^T + bh) * gelu(A·Wg^T + bg), bf16 out [M, FFN]
// Block tile 32(M) x 64(N); wave tile 16x16 with h+gate accumulators.
// TDM path: 3 tiles (A, Wh, Wg) double-buffered.
// ---------------------------------------------------------------------------
__global__ void geglu_gemm(const bf16* __restrict__ A, const bf16* __restrict__ W,
                           const float* __restrict__ bias, bf16* __restrict__ out,
                           int M, int FFN, int K) {
#if HAVE_TDM
  __shared__ bf16 sA[2][32][32];
  __shared__ bf16 sWh[2][64][32];
  __shared__ bf16 sWg[2][64][32];
#else
  __shared__ bf16 sA1[32][32];
  __shared__ bf16 sWh1[64][32];
  __shared__ bf16 sWg1[64][32];
#endif
  int t = threadIdx.x;
  int wave = t >> 5, lane = t & 31;
  int h = lane >> 4, q = lane & 15;
  int wm = wave >> 2, wn = wave & 3;
  int bm = blockIdx.y * 32, bn = blockIdx.x * 64;
  const bf16* Wg = W + (size_t)FFN * K;
  int NT = K >> 5;
  v8f ach = {}, acg = {};
#if HAVE_TDM
  if (wave == 0) {
    tdm_load_tile(LDS_OFF(&sA[0][0][0]),  A  + (size_t)bm * K, 32, 32, M - bm,   K);
    tdm_load_tile(LDS_OFF(&sWh[0][0][0]), W  + (size_t)bn * K, 32, 64, FFN - bn, K);
    tdm_load_tile(LDS_OFF(&sWg[0][0][0]), Wg + (size_t)bn * K, 32, 64, FFN - bn, K);
  }
#endif
  for (int it = 0; it < NT; ++it) {
    int k0 = it << 5;
    __syncthreads();
#if HAVE_TDM
    int buf = it & 1;
    if (wave == 0) {
      if (it + 1 < NT) {
        tdm_load_tile(LDS_OFF(&sA[buf ^ 1][0][0]),  A  + (size_t)bm * K + k0 + 32, 32, 32, M - bm,   K);
        tdm_load_tile(LDS_OFF(&sWh[buf ^ 1][0][0]), W  + (size_t)bn * K + k0 + 32, 32, 64, FFN - bn, K);
        tdm_load_tile(LDS_OFF(&sWg[buf ^ 1][0][0]), Wg + (size_t)bn * K + k0 + 32, 32, 64, FFN - bn, K);
        __builtin_amdgcn_s_wait_tensorcnt(3);
      } else {
        __builtin_amdgcn_s_wait_tensorcnt(0);
      }
    }
    __syncthreads();
    const bf16 (*cA)[32] = sA[buf];
    const bf16 (*cWh)[32] = sWh[buf];
    const bf16 (*cWg)[32] = sWg[buf];
#else
    {
      int idx = t * 4; int r = idx >> 5, c = idx & 31;
      *(uint2*)(&sA1[r][c]) = *(const uint2*)(A + (size_t)(bm + r) * K + k0 + c);
    }
    {
      int idx = t * 8; int r = idx >> 5, c = idx & 31;
      *(uint4*)(&sWh1[r][c]) = *(const uint4*)(W  + (size_t)(bn + r) * K + k0 + c);
      *(uint4*)(&sWg1[r][c]) = *(const uint4*)(Wg + (size_t)(bn + r) * K + k0 + c);
    }
    __syncthreads();
    const bf16 (*cA)[32] = sA1;
    const bf16 (*cWh)[32] = sWh1;
    const bf16 (*cWg)[32] = sWg1;
#endif
    v16bf a  = frag_a(&cA[wm * 16 + q][0], h);
    v16bf b0 = frag_b(&cWh[wn * 16 + q][0], h);
    v16bf b1 = frag_b(&cWg[wn * 16 + q][0], h);
    ach = WMMA_BF16(a, b0, ach);
    acg = WMMA_BF16(a, b1, acg);
  }
  #pragma unroll
  for (int r = 0; r < 8; ++r) {
    int m = bm + wm * 16 + r + 8 * h;
    int n = bn + wn * 16 + q;
    if (m < M) {
      float hv = ach[r] + bias[n];
      float gv = acg[r] + bias[FFN + n];
      float ge = 0.5f * gv * (1.0f + erff(gv * 0.70710678118f));  // exact gelu
      out[(size_t)m * FFN + n] = (bf16)(hv * ge);
    }
  }
}

// ---------------------------------------------------------------------------
// Flash attention. Block = 8 waves, same head; each wave owns a 16-query
// tile. K/V 32x64 blocks staged in LDS (TDM double-buffered; zero-fill past
// Nk comes from TDM OOB semantics). S^T = K·Q^T trick: the WMMA C-layout of
// S^T (lane=query, VGPR-row=key) IS the A-layout of P -> no transpose.
// ---------------------------------------------------------------------------
__global__ void attn_flash(const bf16* __restrict__ Q, const bf16* __restrict__ Kb,
                           const bf16* __restrict__ Vb, bf16* __restrict__ O,
                           int Nk, float scale) {
  const int LD = 1024;                        // H*DH
#if HAVE_TDM
  __shared__ bf16 sK[2][32][64];
  __shared__ bf16 sV[2][32][64];
#else
  __shared__ bf16 sK1[32][64];
  __shared__ bf16 sV1[32][64];
#endif
  int t = threadIdx.x;
  int wave = t >> 5, lane = t & 31;
  int h = lane >> 4, q = lane & 15;
  int head  = blockIdx.y;
  int qbase = (blockIdx.x * 8 + wave) * 16;

  const bf16* qrow = Q + (size_t)(qbase + q) * LD + head * 64;
  v16bf bq0, bq1;
  #pragma unroll
  for (int e = 0; e < 16; ++e) {
    bq0[e] = qrow[16 * h + e];
    bq1[e] = qrow[32 + 16 * h + e];
  }

  v8f ov[4] = {};
  float Mr = -3.0e38f, Lr = 0.f;
  int NT = (Nk + 31) >> 5;
#if HAVE_TDM
  if (wave == 0) {
    tdm_load_tile(LDS_OFF(&sK[0][0][0]), Kb + head * 64, 64, 32, Nk, LD);
    tdm_load_tile(LDS_OFF(&sV[0][0][0]), Vb + head * 64, 64, 32, Nk, LD);
  }
#endif
  for (int it = 0; it < NT; ++it) {
    int kb = it << 5;
    __syncthreads();
#if HAVE_TDM
    int buf = it & 1;
    if (wave == 0) {
      if (it + 1 < NT) {
        int kn = kb + 32;
        tdm_load_tile(LDS_OFF(&sK[buf ^ 1][0][0]), Kb + (size_t)kn * LD + head * 64, 64, 32, Nk - kn, LD);
        tdm_load_tile(LDS_OFF(&sV[buf ^ 1][0][0]), Vb + (size_t)kn * LD + head * 64, 64, 32, Nk - kn, LD);
        __builtin_amdgcn_s_wait_tensorcnt(2);
      } else {
        __builtin_amdgcn_s_wait_tensorcnt(0);
      }
    }
    __syncthreads();
    const bf16 (*cK)[64] = sK[buf];
    const bf16 (*cV)[64] = sV[buf];
#else
    {
      int idx = t * 8; int r = idx >> 6, c = idx & 63;
      int kr = kb + r;
      uint4 kv = {0u, 0u, 0u, 0u}, vv = {0u, 0u, 0u, 0u};
      if (kr < Nk) {
        kv = *(const uint4*)(Kb + (size_t)kr * LD + head * 64 + c);
        vv = *(const uint4*)(Vb + (size_t)kr * LD + head * 64 + c);
      }
      *(uint4*)(&sK1[r][c]) = kv;
      *(uint4*)(&sV1[r][c]) = vv;
    }
    __syncthreads();
    const bf16 (*cK)[64] = sK1;
    const bf16 (*cV)[64] = sV1;
#endif
    v8f s0 = {}, s1 = {};
    #pragma unroll
    for (int tt = 0; tt < 2; ++tt) {          // two 16-key S^T tiles
      const bf16* kp = &cK[tt * 16 + q][0];   // K rows as A operand
      v16bf a0 = frag_a(kp, h);
      v16bf a1 = frag_a(kp + 32, h);
      v8f c = {};
      c = WMMA_BF16(a0, bq0, c);
      c = WMMA_BF16(a1, bq1, c);
      if (tt == 0) s0 = c; else s1 = c;
    }
    float mloc = -3.0e38f;
    #pragma unroll
    for (int r = 0; r < 8; ++r) {
      int k0i = kb + 8 * h + r, k1i = kb + 16 + 8 * h + r;
      float v0 = (k0i < Nk) ? s0[r] * scale : -1.0e30f;
      float v1 = (k1i < Nk) ? s1[r] * scale : -1.0e30f;
      s0[r] = v0; s1[r] = v1;
      mloc = fmaxf(mloc, fmaxf(v0, v1));
    }
    mloc = fmaxf(mloc, __shfl_xor(mloc, 16));
    float Mn = fmaxf(Mr, mloc);
    float alpha = __expf(Mr - Mn);
    float lloc = 0.f;
    #pragma unroll
    for (int r = 0; r < 8; ++r) {
      float p0 = __expf(s0[r] - Mn);
      float p1 = __expf(s1[r] - Mn);
      s0[r] = p0; s1[r] = p1; lloc += p0 + p1;
    }
    lloc += __shfl_xor(lloc, 16);
    Lr = Lr * alpha + lloc;
    Mr = Mn;
    v16bf pa;                                 // P as A operand: in-lane repack
    #pragma unroll
    for (int e = 0; e < 8; ++e) { pa[e] = (bf16)s0[e]; pa[e + 8] = (bf16)s1[e]; }
    #pragma unroll
    for (int r = 0; r < 8; ++r) {
      float ar = __shfl(alpha, r + 8 * h);
      ov[0][r] *= ar; ov[1][r] *= ar; ov[2][r] *= ar; ov[3][r] *= ar;
    }
    #pragma unroll
    for (int dt = 0; dt < 4; ++dt) {
      v16bf bv;
      #pragma unroll
      for (int e = 0; e < 16; ++e) bv[e] = cV[(e < 8) ? (8 * h + e) : (16 + 8 * h + (e - 8))][dt * 16 + q];
      ov[dt] = WMMA_BF16(pa, bv, ov[dt]);
    }
  }
  #pragma unroll
  for (int r = 0; r < 8; ++r) {
    float rl = 1.0f / __shfl(Lr, r + 8 * h);
    size_t m = (size_t)(qbase + r + 8 * h);
    bf16* op = O + m * LD + head * 64 + q;
    op[0]  = (bf16)(ov[0][r] * rl);
    op[16] = (bf16)(ov[1][r] * rl);
    op[32] = (bf16)(ov[2][r] * rl);
    op[48] = (bf16)(ov[3][r] * rl);
  }
}

// ---------------------------------------------------------------------------
// Host orchestration
// ---------------------------------------------------------------------------
extern "C" void kernel_launch(void* const* d_in, const int* in_sizes, int n_in,
                              void* d_out, int out_size, void* d_ws, size_t ws_size,
                              hipStream_t stream) {
  (void)in_sizes; (void)n_in; (void)out_size; (void)ws_size;
  const int N = 2048, D = 1024, H = 16, CN = 77, CD = 768, FF = 4096;
  const int HD = 1024;                   // H*DH
  const float SCALE = 0.125f;            // 64^-0.5

  const float* x0   = (const float*)d_in[0];
  const float* ctx  = (const float*)d_in[1];
  const float* wq1  = (const float*)d_in[2];
  const float* wk1  = (const float*)d_in[3];
  const float* wv1  = (const float*)d_in[4];
  const float* wo1  = (const float*)d_in[5];
  const float* bo1  = (const float*)d_in[6];
  const float* wq2  = (const float*)d_in[7];
  const float* wk2  = (const float*)d_in[8];
  const float* wv2  = (const float*)d_in[9];
  const float* wo2  = (const float*)d_in[10];
  const float* bo2  = (const float*)d_in[11];
  const float* wgg  = (const float*)d_in[12];
  const float* bgg  = (const float*)d_in[13];
  const float* wff  = (const float*)d_in[14];
  const float* bff  = (const float*)d_in[15];
  const float* ln1g = (const float*)d_in[16];
  const float* ln1b = (const float*)d_in[17];
  const float* ln2g = (const float*)d_in[18];
  const float* ln2b = (const float*)d_in[19];
  const float* ln3g = (const float*)d_in[20];
  const float* ln3b = (const float*)d_in[21];
  float* out = (float*)d_out;

  size_t cur = 0;
  auto alloc_bf = [&](size_t n) -> bf16* {
    bf16* p = (bf16*)((char*)d_ws + cur);
    cur += (n * sizeof(bf16) + 255) & ~(size_t)255;
    return p;
  };

  bf16* wq1b = alloc_bf((size_t)HD * D);
  bf16* wk1b = alloc_bf((size_t)HD * D);
  bf16* wv1b = alloc_bf((size_t)HD * D);
  bf16* wo1b = alloc_bf((size_t)D * HD);
  bf16* wq2b = alloc_bf((size_t)HD * D);
  bf16* wk2b = alloc_bf((size_t)HD * CD);
  bf16* wv2b = alloc_bf((size_t)HD * CD);
  bf16* wo2b = alloc_bf((size_t)D * HD);
  bf16* wggb = alloc_bf((size_t)2 * FF * D);
  bf16* wffb = alloc_bf((size_t)D * FF);
  bf16* ctxb = alloc_bf((size_t)CN * CD);
  bf16* xln  = alloc_bf((size_t)N * D);
  bf16* qb   = alloc_bf((size_t)N * HD);
  bf16* kb   = alloc_bf((size_t)N * HD);
  bf16* vb   = alloc_bf((size_t)N * HD);
  bf16* ob   = alloc_bf((size_t)N * HD);
  bf16* hg   = alloc_bf((size_t)N * FF);

  auto cvt = [&](const float* s, bf16* d, int n) {
    cvt_f32_bf16<<<(n + 255) / 256, 256, 0, stream>>>(s, d, n);
  };
  cvt(wq1, wq1b, HD * D);   cvt(wk1, wk1b, HD * D);
  cvt(wv1, wv1b, HD * D);   cvt(wo1, wo1b, D * HD);
  cvt(wq2, wq2b, HD * D);   cvt(wk2, wk2b, HD * CD);
  cvt(wv2, wv2b, HD * CD);  cvt(wo2, wo2b, D * HD);
  cvt(wgg, wggb, 2 * FF * D);
  cvt(wff, wffb, D * FF);
  cvt(ctx, ctxb, CN * CD);

  auto gemm = [&](const bf16* A, const bf16* W, const float* bias, const float* resid,
                  float* oF, bf16* oH, int M, int Nn, int K) {
    dim3 g(Nn / 128, (M + 63) / 64);
    gemm_bf16<<<g, 256, 0, stream>>>(A, W, bias, resid, oF, oH, M, Nn, K);
  };

  // ---- Block 1: self-attention (q, k, v all from layer_norm(x, ln1)) ----
  ln_bf16<<<N, 256, 0, stream>>>(x0, ln1g, ln1b, xln, D);
  gemm(xln, wq1b, nullptr, nullptr, nullptr, qb, N, HD, D);
  gemm(xln, wk1b, nullptr, nullptr, nullptr, kb, N, HD, D);
  gemm(xln, wv1b, nullptr, nullptr, nullptr, vb, N, HD, D);
  attn_flash<<<dim3(N / 16 / 8, H), 256, 0, stream>>>(qb, kb, vb, ob, N, SCALE);
  gemm(ob, wo1b, bo1, x0, out, nullptr, N, D, HD);   // out = attn1 + x

  // ---- Block 2: cross-attention against context ----
  ln_bf16<<<N, 256, 0, stream>>>(out, ln2g, ln2b, xln, D);
  gemm(xln,  wq2b, nullptr, nullptr, nullptr, qb, N,  HD, D);
  gemm(ctxb, wk2b, nullptr, nullptr, nullptr, kb, CN, HD, CD);
  gemm(ctxb, wv2b, nullptr, nullptr, nullptr, vb, CN, HD, CD);
  attn_flash<<<dim3(N / 16 / 8, H), 256, 0, stream>>>(qb, kb, vb, ob, CN, SCALE);
  gemm(ob, wo2b, bo2, out, out, nullptr, N, D, HD);  // out += attn2

  // ---- Block 3: GEGLU FFN ----
  ln_bf16<<<N, 256, 0, stream>>>(out, ln3g, ln3b, xln, D);
  geglu_gemm<<<dim3(FF / 64, N / 32), 256, 0, stream>>>(xln, wggb, bgg, hg, N, FF, D);
  gemm(hg, wffb, bff, out, out, nullptr, N, D, FF);  // out += ffn
}
